// EnhancedAttention_84997402788320
// MI455X (gfx1250) — compile-verified, matrix operand order checked by reading, not run on hardware
//
#include <hip/hip_runtime.h>
#include <hip/hip_bf16.h>

typedef __bf16 bf16;
typedef __attribute__((ext_vector_type(16))) __bf16 v16bf;
typedef __attribute__((ext_vector_type(8)))  __bf16 v8bf;
typedef __attribute__((ext_vector_type(8)))  float  v8f;
typedef __attribute__((ext_vector_type(4)))  unsigned int v4u;
typedef __attribute__((ext_vector_type(8)))  int v8i;
typedef __attribute__((ext_vector_type(4)))  int v4i;

union FragU { v16bf v; v8bf h[2]; };

// Load a 16x32 bf16 WMMA fragment for one lane. Caller passes the lane's base
// pointer (row/col base + khalf). ISA layout: lanes 0-15 hold K {0..7,16..23},
// lanes 16-31 hold K {8..15,24..31}; each half is 8 contiguous bf16 = 16B.
__device__ __forceinline__ v16bf ld_frag16(const bf16* p) {
  FragU u;
  u.h[0] = *(const v8bf*)(p);
  u.h[1] = *(const v8bf*)(p + 16);
  return u.v;
}

__device__ __forceinline__ v8f wmma_bf16(v16bf a, v16bf b, v8f c) {
  // (neg_a, A, neg_b, B, c_mod, C, reuse_a, reuse_b)
  return __builtin_amdgcn_wmma_f32_16x16x32_bf16(false, a, false, b, (short)0, c,
                                                 false, false);
}

// ---------------------------------------------------------------------------
// Tensor Data Mover: DMA a 2D bf16 tile (rows x 32 elems, row stride Kdim
// elems) from global memory into LDS at lds_addr. D# packed per CDNA5 ISA
// ch.8: group0 {count=1, lds_addr, global_addr, type=2}, group1 {data_size=2B,
// tensor_dim0/1, tile_dim0=32, tile_dim1=rows, tensor_dim0_stride=Kdim}.
// ---------------------------------------------------------------------------
__device__ __forceinline__ void tdm_load_tile(unsigned lds_addr, const bf16* gptr,
                                              unsigned rows, unsigned Kdim) {
  unsigned long long ga = (unsigned long long)(size_t)gptr;
  v4u g0;
  g0[0] = 1u;                                            // count=1, is_restore=0
  g0[1] = lds_addr;                                      // LDS byte address
  g0[2] = (unsigned)ga;                                  // global_addr[31:0]
  g0[3] = (unsigned)((ga >> 32) & 0x01ffffffu) | (2u << 30);   // [56:32] | type=2
  v8i g1;
  g1[0] = (int)(1u << 16);                               // data_size=1 -> 2 bytes
  g1[1] = (int)((Kdim & 0xffffu) << 16);                 // tensor_dim0[15:0]
  g1[2] = (int)(((Kdim >> 16) & 0xffffu) | ((rows & 0xffffu) << 16)); // dim0 hi, dim1 lo
  g1[3] = (int)(((rows >> 16) & 0xffffu) | (32u << 16)); // dim1 hi, tile_dim0=32
  g1[4] = (int)(rows & 0xffffu);                         // tile_dim1=rows, tile_dim2=0
  g1[5] = (int)Kdim;                                     // tensor_dim0_stride[31:0]
  g1[6] = 0;                                             // stride hi, dim1_stride lo
  g1[7] = 0;
  v4i z4 = {0, 0, 0, 0};
#if defined(__clang_major__) && (__clang_major__ >= 23)
  v8i z8 = {0, 0, 0, 0, 0, 0, 0, 0};
  __builtin_amdgcn_tensor_load_to_lds(g0, g1, z4, z4, z8, 0);
#else
  __builtin_amdgcn_tensor_load_to_lds(g0, g1, z4, z4, 0);
#endif
}

// ---------------------------------------------------------------------------
// GEMM: C[M,N] = A[M,K] * Bt[N,K]^T, both operands bf16 with K contiguous
// (weights pre-transposed on device). Block tile 128x128, K-step 32,
// 256 threads = 8 wave32 (2x4 wave grid, each wave 64x32 = 4x2 WMMA accums).
// TDM double-buffered LDS staging: wave0 DMAs next tiles while all waves
// run ds_load_b128 fragments + WMMA out of the current buffer.
// ---------------------------------------------------------------------------
template <typename TO>
__global__ __launch_bounds__(256) void gemm_wmma(const bf16* __restrict__ A,
                                                 const bf16* __restrict__ Bt,
                                                 TO* __restrict__ C,
                                                 int M, int N, int K) {
  __shared__ bf16 As[2][128][32];   // [m][k]
  __shared__ bf16 Bs[2][128][32];   // [n][k]

  const int tid  = threadIdx.x;
  const int lane = tid & 31;
  const int wid  = tid >> 5;
  const int wm0  = (wid >> 2) * 64;   // wave M offset (0 or 64)
  const int wn0  = (wid & 3) * 32;    // wave N offset (0,32,64,96)
  const int lr   = lane & 15;
  const int kh   = (lane < 16) ? 0 : 8;
  const int rowoff = (lane < 16) ? 0 : 8;
  const int m0 = blockIdx.y * 128;
  const int n0 = blockIdx.x * 128;

  v8f acc[4][2];
  for (int mi = 0; mi < 4; mi++)
    for (int ni = 0; ni < 2; ni++)
      for (int e = 0; e < 8; e++) acc[mi][ni][e] = 0.f;

  const int nk = K >> 5;
  if (wid == 0) {
    tdm_load_tile((unsigned)(size_t)&As[0][0][0], A + (size_t)m0 * K, 128u, (unsigned)K);
    tdm_load_tile((unsigned)(size_t)&Bs[0][0][0], Bt + (size_t)n0 * K, 128u, (unsigned)K);
  }

  int buf = 0;
  for (int kt = 0; kt < nk; kt++) {
    if (wid == 0) __builtin_amdgcn_s_wait_tensorcnt(0);   // current buf DMA done
    __syncthreads();                                      // publish LDS to all waves
    if (wid == 0 && kt + 1 < nk) {                        // prefetch next tiles
      const int k1 = (kt + 1) << 5;
      tdm_load_tile((unsigned)(size_t)&As[buf ^ 1][0][0],
                    A + (size_t)m0 * K + k1, 128u, (unsigned)K);
      tdm_load_tile((unsigned)(size_t)&Bs[buf ^ 1][0][0],
                    Bt + (size_t)n0 * K + k1, 128u, (unsigned)K);
    }

    v16bf af[4], bfr[2];
#pragma unroll
    for (int mi = 0; mi < 4; mi++) af[mi] = ld_frag16(&As[buf][wm0 + mi * 16 + lr][kh]);
#pragma unroll
    for (int ni = 0; ni < 2; ni++) bfr[ni] = ld_frag16(&Bs[buf][wn0 + ni * 16 + lr][kh]);
#pragma unroll
    for (int mi = 0; mi < 4; mi++)
#pragma unroll
      for (int ni = 0; ni < 2; ni++)
        acc[mi][ni] = wmma_bf16(af[mi], bfr[ni], acc[mi][ni]);

    __syncthreads();   // all reads of buf done before it is DMA-overwritten
    buf ^= 1;
  }

#pragma unroll
  for (int mi = 0; mi < 4; mi++)
#pragma unroll
    for (int ni = 0; ni < 2; ni++)
#pragma unroll
      for (int r = 0; r < 8; r++) {
        const int row = m0 + wm0 + mi * 16 + r + rowoff;
        const int col = n0 + wn0 + ni * 16 + lr;
        C[(size_t)row * N + col] = (TO)acc[mi][ni][r];
      }
}

// ---------------------------------------------------------------------------
// One-time precision/layout prep kernels (tiny vs. the GEMM cost).
// ---------------------------------------------------------------------------
__global__ __launch_bounds__(256) void cvt_f32_bf16(const float* __restrict__ in,
                                                    bf16* __restrict__ out, int n) {
  const int i = (blockIdx.x * 256 + threadIdx.x) * 4;
  if (i + 3 < n) {
    const float4 v = *(const float4*)(in + i);
    out[i + 0] = (bf16)v.x;
    out[i + 1] = (bf16)v.y;
    out[i + 2] = (bf16)v.z;
    out[i + 3] = (bf16)v.w;
  }
}

// Wt[n][k] = (bf16)W[k][n] via 32x32 LDS tile (coalesced both sides).
__global__ __launch_bounds__(256) void transpose_cvt(const float* __restrict__ W,
                                                     bf16* __restrict__ Wt,
                                                     int K, int N) {
  __shared__ float tile[32][33];
  const int k0 = blockIdx.x * 32, n0 = blockIdx.y * 32;
  const int tx = threadIdx.x & 31, ty = threadIdx.x >> 5;   // 32 x 8
#pragma unroll
  for (int j = 0; j < 32; j += 8)
    tile[ty + j][tx] = W[(size_t)(k0 + ty + j) * N + n0 + tx];
  __syncthreads();
#pragma unroll
  for (int j = 0; j < 32; j += 8)
    Wt[(size_t)(n0 + ty + j) * K + k0 + tx] = (bf16)tile[tx][ty + j];
}

// ---------------------------------------------------------------------------
// RoPE + head-major reshuffle. qkv bf16 [B,T,3D] ->
//   Qb,Kb bf16 [B,H,T,DH] (rotated), Vt bf16 [B,H,DH,T] (transposed).
// One thread per (b,t,h,i) with i in [0,32): handles rotation pair (i, i+32).
// ---------------------------------------------------------------------------
__global__ __launch_bounds__(256) void rope_shuffle(const bf16* __restrict__ qkv,
                                                    bf16* __restrict__ Qb,
                                                    bf16* __restrict__ Kb,
                                                    bf16* __restrict__ Vt) {
  unsigned idx = blockIdx.x * blockDim.x + threadIdx.x;   // 2^21 threads
  const int i = idx & 31;
  const int h = (idx >> 5) & 15;
  const int t = (idx >> 9) & 2047;
  const int b = (int)(idx >> 20);

  const bf16* base = qkv + (size_t)(b * 2048 + t) * 3072;
  const float q1 = (float)base[h * 64 + i];
  const float q2 = (float)base[h * 64 + 32 + i];
  const float k1 = (float)base[1024 + h * 64 + i];
  const float k2 = (float)base[1024 + h * 64 + 32 + i];
  const float v1 = (float)base[2048 + h * 64 + i];
  const float v2 = (float)base[2048 + h * 64 + 32 + i];

  const float freq = (float)t * __powf(10000.f, -(float)i * (1.f / 32.f));
  float s, c;
  __sincosf(freq, &s, &c);

  const size_t o = ((size_t)(b * 16 + h) * 2048 + t) * 64;
  Qb[o + i]      = (bf16)(q1 * c - q2 * s);
  Qb[o + 32 + i] = (bf16)(q1 * s + q2 * c);
  Kb[o + i]      = (bf16)(k1 * c - k2 * s);
  Kb[o + 32 + i] = (bf16)(k1 * s + k2 * c);

  const size_t ov = (size_t)(b * 16 + h) * 64;
  Vt[(ov + i) * 2048 + t]      = (bf16)v1;
  Vt[(ov + 32 + i) * 2048 + t] = (bf16)v2;
}

// ---------------------------------------------------------------------------
// Flash attention. Grid (T/64, H, B), 128 threads = 4 wave32; wave w owns
// 16 query rows. Online softmax; P goes through per-wave LDS to convert
// C-layout -> A-fragment layout. AO bf16 [B,T,H,DH] (== [B*T, D] row-major).
// ---------------------------------------------------------------------------
__global__ __launch_bounds__(128) void flash_attn(const bf16* __restrict__ Q,
                                                  const bf16* __restrict__ Kk,
                                                  const bf16* __restrict__ Vt,
                                                  bf16* __restrict__ AO) {
  __shared__ bf16 P[4][16][64];   // per-wave P staging, 8KB

  const int tile = blockIdx.x;
  const int h = blockIdx.y;
  const int b = blockIdx.z;
  const int w = threadIdx.x >> 5;
  const int lane = threadIdx.x & 31;
  const int lr = lane & 15;
  const int kh = (lane < 16) ? 0 : 8;
  const int rowoff = (lane < 16) ? 0 : 8;

  const size_t hb = (size_t)(b * 16 + h) * 2048 * 64;
  const bf16* Qh = Q + hb;
  const bf16* Kh = Kk + hb;
  const bf16* Vh = Vt + hb;   // [dh][t]
  const int q0 = tile * 64 + w * 16;

  v16bf qf[2];
#pragma unroll
  for (int d = 0; d < 2; d++)
    qf[d] = ld_frag16(Qh + (size_t)(q0 + lr) * 64 + d * 32 + kh);

  v8f o[4];
  for (int j = 0; j < 4; j++)
    for (int e = 0; e < 8; e++) o[j][e] = 0.f;
  float rm[8], rl[8];
  for (int r = 0; r < 8; r++) { rm[r] = -3.0e38f; rl[r] = 0.f; }
  const float scale = 0.125f;   // 1/sqrt(64)

  for (int kt = 0; kt <= tile; kt++) {
    const int kbase = kt * 64;

    // S = Q * K^T   (4 key sub-tiles x 2 dh-halves = 8 WMMAs)
    v8f s[4];
    for (int j = 0; j < 4; j++)
      for (int e = 0; e < 8; e++) s[j][e] = 0.f;
#pragma unroll
    for (int ni = 0; ni < 4; ni++) {
      const bf16* kp = Kh + (size_t)(kbase + ni * 16 + lr) * 64;
#pragma unroll
      for (int d = 0; d < 2; d++) {
        v16bf kf = ld_frag16(kp + d * 32 + kh);
        s[ni] = wmma_bf16(qf[d], kf, s[ni]);
      }
    }

    // scale + causal mask
#pragma unroll
    for (int ni = 0; ni < 4; ni++) {
      const int col = kbase + ni * 16 + lr;
#pragma unroll
      for (int r = 0; r < 8; r++) {
        const int row = q0 + r + rowoff;
        const float v = s[ni][r] * scale;
        s[ni][r] = (col <= row) ? v : -3.0e38f;
      }
    }

    // online softmax (row = VGPR index within 16-lane half-groups)
#pragma unroll
    for (int r = 0; r < 8; r++) {
      float m = fmaxf(fmaxf(s[0][r], s[1][r]), fmaxf(s[2][r], s[3][r]));
      m = fmaxf(m, __shfl_xor(m, 1));
      m = fmaxf(m, __shfl_xor(m, 2));
      m = fmaxf(m, __shfl_xor(m, 4));
      m = fmaxf(m, __shfl_xor(m, 8));
      const float mn = fmaxf(rm[r], m);
      const float c = __expf(rm[r] - mn);
      rm[r] = mn;
      float sum = 0.f;
#pragma unroll
      for (int ni = 0; ni < 4; ni++) {
        const float p = __expf(s[ni][r] - mn);
        s[ni][r] = p;
        sum += p;
      }
      sum += __shfl_xor(sum, 1);
      sum += __shfl_xor(sum, 2);
      sum += __shfl_xor(sum, 4);
      sum += __shfl_xor(sum, 8);
      rl[r] = rl[r] * c + sum;
#pragma unroll
      for (int j = 0; j < 4; j++) o[j][r] = o[j][r] * c;
    }

    // stage P (C-layout -> LDS row-major) then reload as A-fragments
#pragma unroll
    for (int ni = 0; ni < 4; ni++)
#pragma unroll
      for (int r = 0; r < 8; r++)
        P[w][r + rowoff][ni * 16 + lr] = (bf16)s[ni][r];
    __syncthreads();   // uniform: all waves run kt = 0..tile

    v16bf pf[2];
#pragma unroll
    for (int d = 0; d < 2; d++) pf[d] = ld_frag16(&P[w][lr][d * 32 + kh]);

    // O += P * V   (4 dh sub-tiles x 2 key-halves = 8 WMMAs)
#pragma unroll
    for (int j = 0; j < 4; j++) {
      const bf16* vp = Vh + (size_t)(j * 16 + lr) * 2048 + kbase;
#pragma unroll
      for (int d = 0; d < 2; d++) {
        v16bf vf = ld_frag16(vp + d * 32 + kh);
        o[j] = wmma_bf16(pf[d], vf, o[j]);
      }
    }
  }

  // normalize + store AO[b][t][h][dh]
#pragma unroll
  for (int j = 0; j < 4; j++)
#pragma unroll
    for (int r = 0; r < 8; r++) {
      const int row = q0 + r + rowoff;
      const float val = o[j][r] / rl[r];
      AO[(size_t)(b * 2048 + row) * 1024 + h * 64 + j * 16 + lr] = (bf16)val;
    }
}

// ---------------------------------------------------------------------------
extern "C" void kernel_launch(void* const* d_in, const int* in_sizes, int n_in,
                              void* d_out, int out_size, void* d_ws, size_t ws_size,
                              hipStream_t stream) {
  const float* x    = (const float*)d_in[0];   // [2,2048,1024]
  const float* Wqkv = (const float*)d_in[1];   // [1024,3072]
  const float* Wout = (const float*)d_in[2];   // [1024,1024]
  float* out = (float*)d_out;                  // [2,2048,1024] f32

  char* ws = (char*)d_ws;
  bf16* xb    = (bf16*)(ws);               // 4096*1024  bf16 =  8388608 B
  bf16* WqkvT = (bf16*)(ws +  8388608);    // 3072*1024  bf16 =  6291456 B
  bf16* WoutT = (bf16*)(ws + 14680064);    // 1024*1024  bf16 =  2097152 B
  bf16* qkv   = (bf16*)(ws + 16777216);    // 4096*3072  bf16 = 25165824 B
  bf16* Qb    = (bf16*)(ws + 41943040);    // 4M bf16 = 8388608 B
  bf16* Kb    = (bf16*)(ws + 50331648);
  bf16* Vt    = (bf16*)(ws + 58720256);
  bf16* AO    = (bf16*)(ws + 67108864);    // total 75497472 B

  // 0) one-time precision/layout prep
  cvt_f32_bf16<<<4096, 256, 0, stream>>>(x, xb, 4096 * 1024);
  dim3 gt1(1024 / 32, 3072 / 32);
  transpose_cvt<<<gt1, 256, 0, stream>>>(Wqkv, WqkvT, 1024, 3072);
  dim3 gt2(1024 / 32, 1024 / 32);
  transpose_cvt<<<gt2, 256, 0, stream>>>(Wout, WoutT, 1024, 1024);

  // 1) QKV projection: [4096,1024] x [1024,3072] -> bf16 qkv (TDM-fed WMMA)
  dim3 g1(3072 / 128, 4096 / 128);
  gemm_wmma<bf16><<<g1, 256, 0, stream>>>(xb, WqkvT, qkv, 4096, 3072, 1024);

  // 2) RoPE + head-major reshuffle (Q,K rotated; V transposed)
  rope_shuffle<<<(2u * 2048u * 16u * 32u) / 256u, 256, 0, stream>>>(qkv, Qb, Kb, Vt);

  // 3) Causal flash attention
  dim3 g3(2048 / 64, 16, 2);
  flash_attn<<<g3, 128, 0, stream>>>(Qb, Kb, Vt, AO);

  // 4) Output projection: [4096,1024] x [1024,1024] -> f32 out
  dim3 g4(1024 / 128, 4096 / 128);
  gemm_wmma<float><<<g4, 256, 0, stream>>>(AO, WoutT, out, 4096, 1024, 1024);
}
